// VectorQuantizer_6227702580023
// MI455X (gfx1250) — matrix operand order repaired and checked by reading, not compiled, not run on hardware
//
#include <hip/hip_runtime.h>
#include <hip/hip_bf16.h>
#include <math.h>

typedef __attribute__((ext_vector_type(16))) __bf16 v16bf;
typedef __attribute__((ext_vector_type(8)))  float  v8f;

#define K_CODES 1024
#define D_DIM   256
#define N_ROWS  32768
#define N_TILES 64   // K / 16
#define N_CHUNK 8    // D / 32
#define TILE_BYTES 16384   // 16 records * 32 lanes * 32B per tile of B operands

// workspace layout (byte offsets)
#define WS_CB    0                       // 64 tiles * 16KB = 1 MiB (codebook, B-operand bf16 hi/lo)
#define WS_EN    (1048576)               // 1024 f32  : ||e_k||^2
#define WS_ZN    (WS_EN  + 4096)         // 32768 f32 : ||z_n||^2
#define WS_PD    (WS_ZN  + 131072)       // 32768 f32 : min_k(||e||^2 - 2 z.e)
#define WS_IDX   (WS_PD  + 131072)       // 32768 i32 : argmin index
#define WS_CNT   (WS_IDX + 131072)       // 1024 u32  : histogram
#define WS_LOSS  (WS_CNT + 4096)         // 1 f32     : sum ||q-z||^2

// output layout (f32 elements)
#define OUT_Q     0
#define OUT_LOSS  8388608
#define OUT_PERP  8388609
#define OUT_IDX   8388610

static __device__ inline __bf16 bf_bits(unsigned short u) {
  union { unsigned short s; __bf16 b; } x; x.s = u; return x.b;
}
static __device__ inline unsigned short f2bf_rne(float f) {
  unsigned int b = __float_as_uint(f);
  unsigned int r = b + 0x7FFFu + ((b >> 16) & 1u);
  return (unsigned short)(r >> 16);
}

// Convert codebook into B-operand records: record ((t*8+c)*2+h), lane ln holds
// 16 bf16: col N = ln&15 (code t*16+N), K (=d) = 32*c + 16*(ln>>4) + j.
__global__ __launch_bounds__(256) void vq_prep_cb(const float* __restrict__ cb,
                                                  char* __restrict__ ws) {
  int gid = blockIdx.x * blockDim.x + threadIdx.x;   // 32768 threads
  int ln = gid & 31;
  int h  = (gid >> 5) & 1;
  int c  = (gid >> 6) & 7;
  int t  = gid >> 9;
  int hb = ln >> 4, n = ln & 15;
  const float* src = cb + (t * 16 + n) * D_DIM + 32 * c + 16 * hb;
  v16bf v;
#pragma unroll
  for (int j = 0; j < 16; j++) {
    float f = src[j];
    unsigned int bits = __float_as_uint(f);
    if (h == 0) {
      v[j] = bf_bits((unsigned short)(bits >> 16));          // truncated hi (exact)
    } else {
      float fhi = __uint_as_float(bits & 0xFFFF0000u);
      v[j] = bf_bits(f2bf_rne(f - fhi));                     // residual lo (RNE)
    }
  }
  ((v16bf*)(ws + WS_CB))[((t * N_CHUNK + c) * 2 + h) * 32 + ln] = v;
}

__global__ __launch_bounds__(256) void vq_prep_en(const float* __restrict__ cb,
                                                  char* __restrict__ ws) {
  int k = blockIdx.x * blockDim.x + threadIdx.x;
  if (k < K_CODES) {
    const float4* p = (const float4*)(cb + k * D_DIM);
    float s = 0.f;
#pragma unroll 4
    for (int j = 0; j < 64; j++) {
      float4 v = p[j];
      s += v.x * v.x + v.y * v.y + v.z * v.z + v.w * v.w;
    }
    ((float*)(ws + WS_EN))[k] = s;
    ((unsigned int*)(ws + WS_CNT))[k] = 0u;   // re-zero every launch (graph replay)
    if (k == 0) *((float*)(ws + WS_LOSS)) = 0.f;
  }
}

// Issue one block-cooperative async tile copy: 256 threads x 64B = 16KB.
// dsaddr/memaddr both get INST_OFFSET added (ISA 10.7 async pseudocode).
#define ASYNC_TILE(T, BUF)                                                    \
  do {                                                                        \
    unsigned _lds = smembase + (unsigned)(BUF)*TILE_BYTES + tid * 64u;        \
    unsigned long long _ga =                                                  \
        cbbase + (unsigned long long)(T)*TILE_BYTES + (unsigned long long)tid * 64ull; \
    asm volatile("global_load_async_to_lds_b128 %0, %1, off\n\t"              \
                 "global_load_async_to_lds_b128 %0, %1, off offset:16\n\t"    \
                 "global_load_async_to_lds_b128 %0, %1, off offset:32\n\t"    \
                 "global_load_async_to_lds_b128 %0, %1, off offset:48"        \
                 :: "v"(_lds), "v"(_ga) : "memory");                          \
  } while (0)

// 8 waves per block, one 16-row z tile per wave (A resident in VGPRs),
// codebook tiles double-buffered through LDS via async-to-LDS DMA.
__global__ __launch_bounds__(256) void vq_argmin(const float* __restrict__ z,
                                                 char* __restrict__ ws) {
  __shared__ __align__(128) char smem[2 * TILE_BYTES];
  const unsigned tid  = threadIdx.x;
  const int lane    = threadIdx.x & 31;
  const int wave    = threadIdx.x >> 5;
  const int rowbase = blockIdx.x * 128 + wave * 16;
  const int ha = lane >> 4, m = lane & 15;
  const float* zr = z + (rowbase + m) * D_DIM;
  const unsigned smembase = (unsigned)(uintptr_t)&smem[0];
  const unsigned long long cbbase = (unsigned long long)(uintptr_t)(ws + WS_CB);

  // Prefetch tile 0 of B into LDS while we load/convert the A tile.
  ASYNC_TILE(0, 0);

  // A-operand tile resident in VGPRs (contiguous k order, same as B records):
  // lane m+16*ha, element j -> d = 32*c + 16*ha + j
  v16bf a_hi[N_CHUNK], a_lo[N_CHUNK];
  float zn = 0.f;
#pragma unroll
  for (int c = 0; c < N_CHUNK; c++) {
    int d0 = 32 * c + 16 * ha;
    float4 va = *(const float4*)(zr + d0);
    float4 vb = *(const float4*)(zr + d0 + 4);
    float4 vc = *(const float4*)(zr + d0 + 8);
    float4 vd = *(const float4*)(zr + d0 + 12);
    float f[16] = {va.x, va.y, va.z, va.w, vb.x, vb.y, vb.z, vb.w,
                   vc.x, vc.y, vc.z, vc.w, vd.x, vd.y, vd.z, vd.w};
#pragma unroll
    for (int j = 0; j < 16; j++) {
      float x = f[j];
      zn += x * x;
      unsigned int bits = __float_as_uint(x);
      a_hi[c][j] = bf_bits((unsigned short)(bits >> 16));
      float fhi = __uint_as_float(bits & 0xFFFF0000u);
      a_lo[c][j] = bf_bits(f2bf_rne(x - fhi));
    }
    __builtin_amdgcn_sched_barrier(0);   // cap live float temporaries per chunk
  }

  const float* en = (const float*)(ws + WS_EN);

  float minv[8];
  int   mini[8];
#pragma unroll
  for (int r = 0; r < 8; r++) { minv[r] = 3.4e38f; mini[r] = 0; }

  asm volatile("s_wait_asynccnt 0" ::: "memory");
  __syncthreads();   // tile 0 resident

  for (int t = 0; t < N_TILES; t++) {
    if (t < N_TILES - 1) ASYNC_TILE(t + 1, (t + 1) & 1);   // prefetch next tile

    const v16bf* bp = (const v16bf*)(smem + (t & 1) * TILE_BYTES) + lane;
    v8f acc = {0.f, 0.f, 0.f, 0.f, 0.f, 0.f, 0.f, 0.f};
#pragma unroll
    for (int c = 0; c < N_CHUNK; c++) {
      v16bf bh = bp[(c * 2 + 0) * 32];   // ds_load_b128 x2
      v16bf bl = bp[(c * 2 + 1) * 32];
      // (zh+zl).(eh+el) ~= zh.eh + zh.el + zl.eh  (zl.el ~ 2^-16, dropped)
      acc = __builtin_amdgcn_wmma_f32_16x16x32_bf16(false, a_hi[c], false, bh,
                                                    (short)0, acc, false, false);
      acc = __builtin_amdgcn_wmma_f32_16x16x32_bf16(false, a_hi[c], false, bl,
                                                    (short)0, acc, false, false);
      acc = __builtin_amdgcn_wmma_f32_16x16x32_bf16(false, a_lo[c], false, bh,
                                                    (short)0, acc, false, false);
    }
    // acc[r] = score[row r + 8*(lane>=16)][col lane&15]
    float e  = en[t * 16 + m];
    int kidx = t * 16 + m;
#pragma unroll
    for (int r = 0; r < 8; r++) {
      float pd = e - 2.f * acc[r];
      if (pd < minv[r]) { minv[r] = pd; mini[r] = kidx; }   // ascending t keeps first min
    }

    asm volatile("s_wait_asynccnt 0" ::: "memory");   // next tile landed
    __syncthreads();                                  // all waves done with buf (t&1)
  }

  // Reduce across the 16 lanes of each half (cols); tie-break = lower index.
#pragma unroll
  for (int off = 1; off < 16; off <<= 1) {
#pragma unroll
    for (int r = 0; r < 8; r++) {
      float ov = __shfl_xor(minv[r], off);
      int   oi = __shfl_xor(mini[r], off);
      if (ov < minv[r] || (ov == minv[r] && oi < mini[r])) { minv[r] = ov; mini[r] = oi; }
    }
  }

  // ||z||^2: lanes m and m+16 each hold half of row m.
  zn += __shfl_xor(zn, 16);
  if (lane < 16) ((float*)(ws + WS_ZN))[rowbase + lane] = zn;

  if (lane == 0 || lane == 16) {
    int rb = rowbase + (lane >> 4) * 8;
    float* pd = (float*)(ws + WS_PD);
    int*   bi = (int*)(ws + WS_IDX);
#pragma unroll
    for (int r = 0; r < 8; r++) { pd[rb + r] = minv[r]; bi[rb + r] = mini[r]; }
  }
}

// Gather quantized rows, histogram, loss partial sums, index output.
__global__ __launch_bounds__(256) void vq_gather(const float* __restrict__ cb,
                                                 float* __restrict__ out,
                                                 char* __restrict__ ws) {
  int gid = blockIdx.x * 256 + threadIdx.x;   // N_ROWS*64 threads (float4 granularity)
  int n = gid >> 6, q = gid & 63;
  int idx = ((const int*)(ws + WS_IDX))[n];
  float4 v = ((const float4*)cb)[idx * 64 + q];
  ((float4*)(out + OUT_Q))[(size_t)n * 64 + q] = v;   // straight-through fwd == quantized

  float contrib = 0.f;
  if (q == 0) {
    atomicAdd((unsigned int*)(ws + WS_CNT) + idx, 1u);
    contrib = ((const float*)(ws + WS_ZN))[n] + ((const float*)(ws + WS_PD))[n];
    out[OUT_IDX + n] = (float)idx;
  }
  __shared__ float red[256];
  red[threadIdx.x] = contrib;
  __syncthreads();
  for (int s = 128; s > 0; s >>= 1) {
    if (threadIdx.x < s) red[threadIdx.x] += red[threadIdx.x + s];
    __syncthreads();
  }
  if (threadIdx.x == 0) atomicAdd((float*)(ws + WS_LOSS), red[0]);
}

__global__ __launch_bounds__(256) void vq_finalize(float* __restrict__ out,
                                                   const char* __restrict__ ws) {
  __shared__ float red[256];
  const unsigned int* cnt = (const unsigned int*)(ws + WS_CNT);
  float s = 0.f;
  for (int k = threadIdx.x; k < K_CODES; k += 256) {
    float p = (float)cnt[k] * (1.0f / (float)N_ROWS);
    s += p * logf(p + 1e-10f);
  }
  red[threadIdx.x] = s;
  __syncthreads();
  for (int t = 128; t > 0; t >>= 1) {
    if (threadIdx.x < t) red[threadIdx.x] += red[threadIdx.x + t];
    __syncthreads();
  }
  if (threadIdx.x == 0) {
    // q_latent_loss == e_latent_loss forward => loss = 1.25 * mean((q-z)^2)
    out[OUT_LOSS] = 1.25f * (*(const float*)(ws + WS_LOSS)) / 8388608.0f;
    out[OUT_PERP] = expf(-red[0]);
  }
}

extern "C" void kernel_launch(void* const* d_in, const int* in_sizes, int n_in,
                              void* d_out, int out_size, void* d_ws, size_t ws_size,
                              hipStream_t stream) {
  const float* z  = (const float*)d_in[0];     // [32,1024,256]
  const float* cb = (const float*)d_in[1];     // [1024,256]
  float* out = (float*)d_out;
  char*  ws  = (char*)d_ws;

  vq_prep_cb<<<128, 256, 0, stream>>>(cb, ws);            // 32768 threads
  vq_prep_en<<<4, 256, 0, stream>>>(cb, ws);              // 1024 threads
  vq_argmin <<<N_ROWS / 128, 256, 0, stream>>>(z, ws);    // 256 blocks x 8 waves
  vq_gather <<<(N_ROWS * 64) / 256, 256, 0, stream>>>(cb, out, ws);
  vq_finalize<<<1, 256, 0, stream>>>(out, ws);
}
